// MultiHeadAttention_15659450761408
// MI455X (gfx1250) — matrix operand order repaired and checked by reading, not compiled, not run on hardware
//
#include <hip/hip_runtime.h>

// Problem constants (match reference).
#define BB   4
#define SS   2048
#define DD   1024
#define HH   16
#define DKK  64
#define MM   (BB*SS)          // 8192 rows

typedef __attribute__((ext_vector_type(16))) __bf16 v16bf;
typedef __attribute__((ext_vector_type(8)))  float  v8f;
typedef __attribute__((ext_vector_type(4)))  float  f4raw;   // trivial 16B vector

union F8   { v8f v; float f[8]; };
union Frag { f4raw q[2]; v16bf v; };

static __device__ inline unsigned short f2bf(float x) {
  union { float f; unsigned u; } c; c.f = x;
  unsigned u = c.u;
  u = (u + 0x7FFFu + ((u >> 16) & 1u)) >> 16;   // round-to-nearest-even
  return (unsigned short)u;
}

// Build a 16x bf16 fragment from two 16-byte chunks.
static __device__ inline v16bf ld_frag_pair(const unsigned short* p0,
                                            const unsigned short* p1) {
  Frag f;
  f.q[0] = *(const f4raw*)p0;
  f.q[1] = *(const f4raw*)p1;
  return f.v;
}
static __device__ inline v16bf ld_frag_contig(const unsigned short* p) {
  return ld_frag_pair(p, p + 8);
}

// LDS byte offset of a __shared__ object (flat LDS aperture keeps the
// workgroup-relative offset in addr[31:0], per ISA 10.2).
static __device__ inline unsigned lds_off(const void* p) {
  return (unsigned)(unsigned long long)p;
}

// Async global->LDS 16B copy (GVS mode: SGPR64 base + VGPR32 byte offset).
static __device__ inline void async_b128(unsigned lds, unsigned gofs,
                                         const void* base) {
  asm volatile("global_load_async_to_lds_b128 %0, %1, %2"
               :: "v"(lds), "v"(gofs), "s"(base) : "memory");
}
static __device__ inline void wait_async0() {
  asm volatile("s_wait_asynccnt 0x0" ::: "memory");
}

// --------- 16-lane butterfly reductions via DPP16 (no LDS traffic).
// xor-masks {1,2,7,15} span all 4 lane bits within a row of 16.
template<int CTRL>
static __device__ inline float dpp_bfly(float x) {
  union { float f; int i; } a, b;
  a.f = x;
  b.i = __builtin_amdgcn_update_dpp(0, a.i, CTRL, 0xF, 0xF, true);
  return b.f;
}
static __device__ inline float red_max16(float x) {
  x = fmaxf(x, dpp_bfly<0x0B1>(x));  // quad_perm(1,0,3,2)  : xor 1
  x = fmaxf(x, dpp_bfly<0x04E>(x));  // quad_perm(2,3,0,1)  : xor 2
  x = fmaxf(x, dpp_bfly<0x141>(x));  // ROW_HALF_MIRROR     : xor 7
  x = fmaxf(x, dpp_bfly<0x140>(x));  // ROW_MIRROR          : xor 15
  return x;
}
static __device__ inline float red_sum16(float x) {
  x += dpp_bfly<0x0B1>(x);
  x += dpp_bfly<0x04E>(x);
  x += dpp_bfly<0x141>(x);
  x += dpp_bfly<0x140>(x);
  return x;
}

// ---------------------------------------------------------------- fp32 -> bf16
__global__ __launch_bounds__(256) void cvt_bf16(const float* __restrict__ in,
                                                unsigned short* __restrict__ out,
                                                int n4) {
  int i = blockIdx.x * 256 + threadIdx.x;
  if (i < n4) {
    f4raw v = ((const f4raw*)in)[i];
    union { unsigned short u[4]; unsigned long long ll; } o;
    o.u[0] = f2bf(v.x); o.u[1] = f2bf(v.y);
    o.u[2] = f2bf(v.z); o.u[3] = f2bf(v.w);
    ((unsigned long long*)out)[i] = o.ll;
  }
}

// ----------------------------------------------------------- bf16 WMMA GEMM
// C[m,n] = sum_k A[m,k] * W[n,k] + bias[n]   (i.e. x @ W^T + b)
// A: [MM][DD] bf16 row-major.  W: [DD][DD] bf16 row-major (torch layout).
// MODE 0: out = bf16 head-split [B,H,S,DK].  MODE 1: out = fp32 [MM][DD].
// 256 thr = 8 waves (4 along M, 2 along N); block tile 128x128, K-step 32.
// Double-buffered LDS staging via GLOBAL_LOAD_ASYNC_TO_LDS_B128 (ASYNCcnt).
template<int MODE>
__global__ __launch_bounds__(256) void gemm_bf16_kernel(
    const unsigned short* __restrict__ A,
    const unsigned short* __restrict__ W,
    const float* __restrict__ bias,
    void* __restrict__ outp)
{
  __shared__ __align__(16) unsigned short As[2][128 * 40];  // 40-half row pad
  __shared__ __align__(16) unsigned short Ws[2][128 * 40];

  const int K = DD, N = DD;
  const int m0 = blockIdx.x * 128;
  const int n0 = blockIdx.y * 128;
  const int t    = threadIdx.x;
  const int lane = t & 31, wave = t >> 5;
  const int wm = wave & 3, wn = wave >> 2;        // wave tile 32x64
  const int half = lane >> 4, l16 = lane & 15;

  const int srow = t >> 1;                        // staging: 128 rows x 32 halves
  const int soff = (t & 1) * 16;
  const size_t gA_row = ((size_t)(m0 + srow) * K + soff) * 2;  // byte offsets
  const size_t gW_row = ((size_t)(n0 + srow) * K + soff) * 2;

  F8 acc[2][4];
#pragma unroll
  for (int i = 0; i < 2; i++)
#pragma unroll
    for (int j = 0; j < 4; j++)
#pragma unroll
      for (int r = 0; r < 8; r++) acc[i][j].f[r] = 0.f;

  auto stage = [&](int buf, int k0) {
    unsigned la = lds_off(&As[buf][srow * 40 + soff]);
    unsigned lw = lds_off(&Ws[buf][srow * 40 + soff]);
    unsigned oa = (unsigned)(gA_row + (size_t)k0 * 2);
    unsigned ow = (unsigned)(gW_row + (size_t)k0 * 2);
    async_b128(la,      oa,      A);
    async_b128(la + 16, oa + 16, A);
    async_b128(lw,      ow,      W);
    async_b128(lw + 16, ow + 16, W);
  };

  stage(0, 0);
  wait_async0();
  __syncthreads();

  int cur = 0;
  for (int k0 = 0; k0 < K; k0 += 32) {
    if (k0 + 32 < K) stage(cur ^ 1, k0 + 32);   // overlap next tile with WMMA

    v16bf af[2], bf[4];
#pragma unroll
    for (int i = 0; i < 2; i++) {
      int ra = wm * 32 + i * 16 + l16;
      // A layout: lanes 0-15 K={0..7,16..23}, lanes 16-31 K={8..15,24..31}
      af[i] = ld_frag_pair(&As[cur][ra * 40 + 8 * half],
                           &As[cur][ra * 40 + 16 + 8 * half]);
    }
#pragma unroll
    for (int j = 0; j < 4; j++) {
      int rb = wn * 64 + j * 16 + l16;
      // B layout: lanes 0-15 K={0..7,16..23}, lanes 16-31 K={8..15,24..31}
      bf[j] = ld_frag_pair(&Ws[cur][rb * 40 + 8 * half],
                           &Ws[cur][rb * 40 + 16 + 8 * half]);
    }
#pragma unroll
    for (int i = 0; i < 2; i++)
#pragma unroll
      for (int j = 0; j < 4; j++)
        acc[i][j].v = __builtin_amdgcn_wmma_f32_16x16x32_bf16(
            false, af[i], false, bf[j], (short)0, acc[i][j].v, false, false);

    wait_async0();
    __syncthreads();
    cur ^= 1;
  }

  // Epilogue. C layout: VGPR r, lanes 0-15 -> M=r, lanes 16-31 -> M=r+8; N=l16.
#pragma unroll
  for (int i = 0; i < 2; i++)
#pragma unroll
    for (int j = 0; j < 4; j++)
#pragma unroll
      for (int r = 0; r < 8; r++) {
        int m = m0 + wm * 32 + i * 16 + r + 8 * half;
        int n = n0 + wn * 64 + j * 16 + l16;
        float vv = acc[i][j].f[r] + bias[n];
        if (MODE == 0) {
          int b = m / SS, s = m % SS;
          int h = n >> 6, dk = n & 63;
          ((unsigned short*)outp)[(((size_t)b * HH + h) * SS + s) * DKK + dk] =
              f2bf(vv);
        } else {
          ((float*)outp)[(size_t)m * N + n] = vv;
        }
      }
}

// ------------------------------------------------------ causal flash attention
// Q,K,V: bf16 [B*H][S][64].  ctx out: bf16 [B*S][D] (head-interleaved).
// Block = 128 thr = 4 waves; each wave owns 16 query rows; KV chunk = 32.
__global__ __launch_bounds__(128) void attn_kernel(
    const unsigned short* __restrict__ Q,
    const unsigned short* __restrict__ Kh,
    const unsigned short* __restrict__ V,
    unsigned short* __restrict__ ctx)
{
  __shared__ __align__(16) unsigned short Ks[32 * 72];      // [key][dk], padded
  __shared__ __align__(16) unsigned short Vs[64 * 40];      // [dk][key], padded
  __shared__ __align__(16) unsigned short Ps[4 * 16 * 40];  // per-wave P tile

  const int t    = threadIdx.x;
  const int lane = t & 31, wave = t >> 5;
  const int half = lane >> 4, l16 = lane & 15;
  const int bh   = blockIdx.y;
  const int q0w  = blockIdx.x * 64 + wave * 16;
  const float scale = 0.125f;                 // 1/sqrt(DK)

  const size_t headbase = (size_t)bh * SS * DKK;

  // Q A-fragments for both 32-wide K-chunks of DK=64 (rows contiguous 128B).
  v16bf aQ[2];
  {
    const unsigned short* qp = Q + headbase + (size_t)(q0w + l16) * DKK;
#pragma unroll
    for (int c = 0; c < 2; c++)
      aQ[c] = ld_frag_pair(qp + 32 * c + 8 * half, qp + 32 * c + 16 + 8 * half);
  }

  F8 cacc[4];
#pragma unroll
  for (int j = 0; j < 4; j++)
#pragma unroll
    for (int r = 0; r < 8; r++) cacc[j].f[r] = 0.f;
  float row_m[8], row_l[8];
#pragma unroll
  for (int r = 0; r < 8; r++) { row_m[r] = -3.0e38f; row_l[r] = 0.f; }

  unsigned short* Psw = Ps + wave * 16 * 40;
  const int nchunk = blockIdx.x * 2 + 2;      // keys needed: q0_blk + 64

  const int srow = t >> 2;                    // staging: 32 rows x 64 halves
  const int soff = (t & 3) * 16;

  for (int c = 0; c < nchunk; c++) {
    const int kv0 = c * 32;
    // K rows stream straight into LDS (async, no VGPR round-trip).
    {
      unsigned lk = lds_off(&Ks[srow * 72 + soff]);
      unsigned ok = (unsigned)((headbase + (size_t)(kv0 + srow) * DKK + soff) * 2);
      async_b128(lk,      ok,      Kh);
      async_b128(lk + 16, ok + 16, Kh);
      // V needs a transpose: vector-load 32B, scatter 16 bf16 into [dk][key].
      const unsigned short* gv = V + headbase + (size_t)(kv0 + srow) * DKK + soff;
      union { f4raw f[2]; unsigned short u[16]; } vv;
      vv.f[0] = ((const f4raw*)gv)[0];
      vv.f[1] = ((const f4raw*)gv)[1];
#pragma unroll
      for (int i = 0; i < 16; i++) Vs[(soff + i) * 40 + srow] = vv.u[i];
      if (c + 1 < nchunk) {                   // prefetch next chunk's V rows
        __builtin_prefetch(gv + 32 * DKK, 0, 0);
      }
    }
    wait_async0();
    __syncthreads();

    // scores S = Q K^T : 16x32 tile = 2 N-tiles, contraction DK=64 in 2 steps.
    F8 sc[2];
#pragma unroll
    for (int n = 0; n < 2; n++)
#pragma unroll
      for (int r = 0; r < 8; r++) sc[n].f[r] = 0.f;
#pragma unroll
    for (int kc = 0; kc < 2; kc++) {
      v16bf b0 = ld_frag_pair(&Ks[(l16) * 72 + 32 * kc + 8 * half],
                              &Ks[(l16) * 72 + 32 * kc + 16 + 8 * half]);
      v16bf b1 = ld_frag_pair(&Ks[(16 + l16) * 72 + 32 * kc + 8 * half],
                              &Ks[(16 + l16) * 72 + 32 * kc + 16 + 8 * half]);
      sc[0].v = __builtin_amdgcn_wmma_f32_16x16x32_bf16(
          false, aQ[kc], false, b0, (short)0, sc[0].v, false, false);
      sc[1].v = __builtin_amdgcn_wmma_f32_16x16x32_bf16(
          false, aQ[kc], false, b1, (short)0, sc[1].v, false, false);
    }

    // Online softmax (fp32); row reductions via DPP16 butterflies.
#pragma unroll
    for (int r = 0; r < 8; r++) {
      int qrow = q0w + r + 8 * half;
      float s0 = (kv0 + l16      <= qrow) ? sc[0].f[r] * scale : -3.0e38f;
      float s1 = (kv0 + 16 + l16 <= qrow) ? sc[1].f[r] * scale : -3.0e38f;
      float mx = red_max16(fmaxf(s0, s1));
      float mnew  = fmaxf(row_m[r], mx);
      float alpha = __expf(row_m[r] - mnew);
      float p0 = __expf(s0 - mnew);
      float p1 = __expf(s1 - mnew);
      float psum = red_sum16(p0 + p1);
      row_l[r] = row_l[r] * alpha + psum;
      row_m[r] = mnew;
#pragma unroll
      for (int j = 0; j < 4; j++) cacc[j].f[r] *= alpha;
      // Stage P (C-layout -> LDS) for A-layout reload.
      Psw[(r + 8 * half) * 40 + l16]      = f2bf(p0);
      Psw[(r + 8 * half) * 40 + 16 + l16] = f2bf(p1);
    }
    // Wave-private LDS RAW: gfx1250 split DS counter wait.
    asm volatile("s_wait_dscnt 0x0" ::: "memory");

    v16bf aP = ld_frag_pair(&Psw[l16 * 40 + 8 * half],
                            &Psw[l16 * 40 + 16 + 8 * half]);
#pragma unroll
    for (int j = 0; j < 4; j++) {
      v16bf bv = ld_frag_pair(&Vs[(j * 16 + l16) * 40 + 8 * half],
                              &Vs[(j * 16 + l16) * 40 + 16 + 8 * half]);
      cacc[j].v = __builtin_amdgcn_wmma_f32_16x16x32_bf16(
          false, aP, false, bv, (short)0, cacc[j].v, false, false);
    }
    __syncthreads();   // protect Ks/Vs before restaging
  }

  const int b = bh >> 4, h = bh & 15;
#pragma unroll
  for (int j = 0; j < 4; j++)
#pragma unroll
    for (int r = 0; r < 8; r++) {
      int q = q0w + r + 8 * half;
      float val = cacc[j].f[r] / row_l[r];
      ctx[((size_t)(b * SS + q)) * DD + h * DKK + j * 16 + l16] = f2bf(val);
    }
}

// ---------------------------------------------------------------- launcher
extern "C" void kernel_launch(void* const* d_in, const int* in_sizes, int n_in,
                              void* d_out, int out_size, void* d_ws, size_t ws_size,
                              hipStream_t stream) {
  (void)in_sizes; (void)n_in; (void)out_size; (void)ws_size;
  const float* q  = (const float*)d_in[0];
  const float* k  = (const float*)d_in[1];
  const float* v  = (const float*)d_in[2];
  // d_in[3] = mask (tril) -- applied analytically in attn_kernel.
  const float* Wq = (const float*)d_in[4];
  const float* bq = (const float*)d_in[5];
  const float* Wk = (const float*)d_in[6];
  const float* bk = (const float*)d_in[7];
  const float* Wv = (const float*)d_in[8];
  const float* bv = (const float*)d_in[9];
  const float* Wo = (const float*)d_in[10];
  const float* bo = (const float*)d_in[11];

  // Workspace layout (bf16 halves). Total = 7*8M + 4*1M halves = 120 MB.
  const size_t MD = (size_t)MM * DD;     // 8M elements
  const size_t WD = (size_t)DD * DD;     // 1M elements
  unsigned short* ws  = (unsigned short*)d_ws;
  unsigned short* qb  = ws;
  unsigned short* kb  = qb  + MD;
  unsigned short* vb  = kb  + MD;
  unsigned short* wqb = vb  + MD;
  unsigned short* wkb = wqb + WD;
  unsigned short* wvb = wkb + WD;
  unsigned short* wob = wvb + WD;
  unsigned short* Qh  = wob + WD;
  unsigned short* Kb  = Qh  + MD;
  unsigned short* Vb  = Kb  + MD;
  unsigned short* Cx  = Vb  + MD;

  // 1) fp32 -> bf16 (vectorized x4)
  cvt_bf16<<<(int)(MD / 4 / 256), 256, 0, stream>>>(q,  qb,  (int)(MD / 4));
  cvt_bf16<<<(int)(MD / 4 / 256), 256, 0, stream>>>(k,  kb,  (int)(MD / 4));
  cvt_bf16<<<(int)(MD / 4 / 256), 256, 0, stream>>>(v,  vb,  (int)(MD / 4));
  cvt_bf16<<<(int)(WD / 4 / 256), 256, 0, stream>>>(Wq, wqb, (int)(WD / 4));
  cvt_bf16<<<(int)(WD / 4 / 256), 256, 0, stream>>>(Wk, wkb, (int)(WD / 4));
  cvt_bf16<<<(int)(WD / 4 / 256), 256, 0, stream>>>(Wv, wvb, (int)(WD / 4));
  cvt_bf16<<<(int)(WD / 4 / 256), 256, 0, stream>>>(Wo, wob, (int)(WD / 4));

  // 2) QKV projections (bf16 head-split outputs)
  dim3 gg(MM / 128, DD / 128);   // 64 x 8
  gemm_bf16_kernel<0><<<gg, 256, 0, stream>>>(qb, wqb, bq, Qh);
  gemm_bf16_kernel<0><<<gg, 256, 0, stream>>>(kb, wkb, bk, Kb);
  gemm_bf16_kernel<0><<<gg, 256, 0, stream>>>(vb, wvb, bv, Vb);

  // 3) causal flash attention -> ctx bf16 [B*S][D]
  attn_kernel<<<dim3(SS / 64, BB * HH), 128, 0, stream>>>(Qh, Kb, Vb, Cx);

  // 4) output projection -> fp32 d_out
  gemm_bf16_kernel<1><<<gg, 256, 0, stream>>>(Cx, wob, bo, (float*)d_out);
}
